// EGraphSAGE_6949257085052
// MI455X (gfx1250) — compile-verified, matrix operand order checked by reading, not run on hardware
//
#include <hip/hip_runtime.h>

// ---------------------------------------------------------------------------
// E-GraphSAGE forward for MI455X (gfx1250), wave32 + V_WMMA_F32_16X16X32_BF16.
// Weights staged once per block into LDS as bf16 fragments (compiler hoists
// the per-lane fragment into VGPRs for the whole tile loop); each wave runs a
// grid-stride loop over 16-row tiles (wave-uniform -> EXEC all-ones at WMMA).
// Full tiles take an unguarded scatter/store path; only the single possible
// tail tile pays per-element EXEC masking.
// ---------------------------------------------------------------------------

typedef __attribute__((ext_vector_type(16))) __bf16 v16bf;
typedef __attribute__((ext_vector_type(8)))  float  v8f;

__device__ __forceinline__ v8f wmma_bf16(v16bf a, v16bf b, v8f c) {
  // 8 args: (neg_a, A, neg_b, B, c_mod, C, reuse_a, reuse_b)
  return __builtin_amdgcn_wmma_f32_16x16x32_bf16(false, a, false, b,
                                                 (short)0, c, false, false);
}

// Build one lane's half of a 16x32 bf16 A/B fragment from a contiguous f32
// row. klo = full-K coordinate of this lane's first 8-element chunk
// (ISA 7.12.2: lanes 0-15 own K [0,8)+[16,24), lanes 16-31 own K [8,16)+[24,32)).
__device__ __forceinline__ v16bf mk_frag(const float* __restrict__ rp, int klo,
                                         float scale) {
  v16bf a;
#pragma unroll
  for (int i = 0; i < 8; ++i) a[i] = (__bf16)(rp[klo + i] * scale);
#pragma unroll
  for (int i = 0; i < 8; ++i) a[8 + i] = (__bf16)(rp[klo + 16 + i] * scale);
  return a;
}

// Stage a [64 out][128 in] f32 weight matrix into LDS as bf16 WMMA-B
// fragments: Wlds[nt][ks][lane] is the v16bf this lane feeds WMMA for
// N-tile nt, K-step ks.  4*4*32*32B = 16 KB.
__device__ __forceinline__ void stage_W(v16bf (*Wlds)[4][32],
                                        const float* __restrict__ W) {
  for (int s = threadIdx.x; s < 4 * 4 * 32; s += blockDim.x) {
    const int nt = s >> 7, ks = (s >> 5) & 3, ln = s & 31;
    const int half = ln >> 4, mrow = ln & 15;
    const float* wrow = W + (long)(nt * 16 + mrow) * 128;  // B col n = W row n
    Wlds[nt][ks][ln] = mk_frag(wrow, ks * 32 + half * 8, 1.0f);
  }
  __syncthreads();
}

// ---------------------------------------------------------------------------
// Kernel 0: zero sums[N*64] and cnt[N] (contiguous region in workspace)
// ---------------------------------------------------------------------------
__global__ void k_zero(float4* __restrict__ p4, long n4,
                       float* __restrict__ tailp, int tail) {
  long i = (long)blockIdx.x * blockDim.x + threadIdx.x;
  if (i < n4) p4[i] = make_float4(0.f, 0.f, 0.f, 0.f);
  if (blockIdx.x == 0 && (int)threadIdx.x < tail) tailp[threadIdx.x] = 0.f;
}

// ---------------------------------------------------------------------------
// Kernel 1: msg = concat(x[src], edge_attr) @ W_msg^T + b ; scatter-add by dst
// One wave per 16-edge tile (grid-stride). K = 128, N = 64.
// ---------------------------------------------------------------------------
__global__ void __launch_bounds__(256)
k_msg_scatter(const float* __restrict__ x,
              const long long* __restrict__ ei,   // [2][E] int64
              const float* __restrict__ ea,       // [E][64]
              const float* __restrict__ Wm,       // [64][128] (out,in)
              const float* __restrict__ bm,       // [64]
              float* __restrict__ sums,           // [N][64]
              float* __restrict__ cnt,            // [N]
              int E) {
  __shared__ v16bf Wlds[4][4][32];
  stage_W(Wlds, Wm);

  const int lane = (int)threadIdx.x & 31;
  const int half = lane >> 4;                 // K-chunk / D-row half
  const int mrow = lane & 15;                 // A row / D column within tile
  const int wid  = blockIdx.x * (blockDim.x >> 5) + ((int)threadIdx.x >> 5);
  const int wstride = gridDim.x * (blockDim.x >> 5);
  const int ntiles  = (E + 15) >> 4;

  float bias[4];                              // tile-loop invariant per lane
#pragma unroll
  for (int nt = 0; nt < 4; ++nt) bias[nt] = bm[nt * 16 + mrow];

  for (int tile = wid; tile < ntiles; tile += wstride) {  // wave-uniform loop
    const long base = (long)tile * 16;

    long eA = base + mrow;
    if (eA >= E) eA = E - 1;                  // clamp; garbage rows masked below
    const int srcA = (int)ei[eA];
    const int dstA = (int)ei[(long)E + eA];   // lane m holds dst of edge base+m
    const float* xrow  = x  + (long)srcA * 64;
    const float* earow = ea + eA * 64;

    // A fragments over K = 0..127 (x: 0..63, edge_attr: 64..127)
    v16bf afrag[4];
#pragma unroll
    for (int ks = 0; ks < 4; ++ks) {
      const float* rp = (ks < 2) ? xrow : (earow - 64);   // full-K indexing
      afrag[ks] = mk_frag(rp, ks * 32 + half * 8, 1.0f);
    }

    // Scatter targets for the 8 D rows this lane owns: cross-lane permute,
    // no memory traffic (lane r holds dst[base+r]).
    int dsts[8];
#pragma unroll
    for (int v = 0; v < 8; ++v) dsts[v] = __shfl(dstA, half * 8 + v, 32);

    // D = A @ B for all four 16-wide N tiles (EXEC all-ones here)
    v8f accs[4];
#pragma unroll
    for (int nt = 0; nt < 4; ++nt) {
      v8f acc = {};
#pragma unroll
      for (int ks = 0; ks < 4; ++ks)
        acc = wmma_bf16(afrag[ks], Wlds[nt][ks][lane], acc);
      accs[nt] = acc;
    }

    if (base + 16 <= E) {                     // full tile: unguarded scatter
      if (half == 0) atomicAdd(&cnt[dstA], 1.0f);
#pragma unroll
      for (int nt = 0; nt < 4; ++nt) {
        const int   n = nt * 16 + mrow;
        const float b = bias[nt];
#pragma unroll
        for (int v = 0; v < 8; ++v)
          atomicAdd(&sums[(long)dsts[v] * 64 + n], accs[nt][v] + b);
      }
    } else {                                  // tail tile (at most one)
      if (half == 0 && base + mrow < E) atomicAdd(&cnt[dstA], 1.0f);
#pragma unroll
      for (int nt = 0; nt < 4; ++nt) {
        const int   n = nt * 16 + mrow;
        const float b = bias[nt];
#pragma unroll
        for (int v = 0; v < 8; ++v)
          if (base + half * 8 + v < E)
            atomicAdd(&sums[(long)dsts[v] * 64 + n], accs[nt][v] + b);
      }
    }
  }
}

// ---------------------------------------------------------------------------
// Kernel 2: h = relu(concat(x, sums/max(cnt,1)) @ W_apply^T + b)
// One wave per 16-node tile (grid-stride).
// ---------------------------------------------------------------------------
__global__ void __launch_bounds__(256)
k_apply(const float* __restrict__ x,
        const float* __restrict__ sums,
        const float* __restrict__ cnt,
        const float* __restrict__ Wa,          // [64][128]
        const float* __restrict__ ba,          // [64]
        float* __restrict__ h,                 // [N][64]
        int N) {
  __shared__ v16bf Wlds[4][4][32];
  stage_W(Wlds, Wa);

  const int lane = (int)threadIdx.x & 31;
  const int half = lane >> 4;
  const int mrow = lane & 15;
  const int wid  = blockIdx.x * (blockDim.x >> 5) + ((int)threadIdx.x >> 5);
  const int wstride = gridDim.x * (blockDim.x >> 5);
  const int ntiles  = (N + 15) >> 4;

  float bias[4];
#pragma unroll
  for (int nt = 0; nt < 4; ++nt) bias[nt] = ba[nt * 16 + mrow];

  for (int tile = wid; tile < ntiles; tile += wstride) {  // wave-uniform loop
    const long base = (long)tile * 16;
    long node = base + mrow;
    if (node >= N) node = N - 1;
    const float  invc = 1.0f / fmaxf(cnt[node], 1.0f);    // scatter-mean
    const float* xrow = x + node * 64;
    const float* srow = sums + node * 64;

    v16bf afrag[4];
#pragma unroll
    for (int ks = 0; ks < 4; ++ks) {
      const bool   isX = (ks < 2);
      const float* rp  = isX ? xrow : (srow - 64);
      afrag[ks] = mk_frag(rp, ks * 32 + half * 8, isX ? 1.0f : invc);
    }

    v8f accs[4];
#pragma unroll
    for (int nt = 0; nt < 4; ++nt) {
      v8f acc = {};
#pragma unroll
      for (int ks = 0; ks < 4; ++ks)
        acc = wmma_bf16(afrag[ks], Wlds[nt][ks][lane], acc);
      accs[nt] = acc;
    }

    if (base + 16 <= N) {                     // full tile: unguarded stores
#pragma unroll
      for (int nt = 0; nt < 4; ++nt) {
        const int n = nt * 16 + mrow;
#pragma unroll
        for (int v = 0; v < 8; ++v)
          h[(base + half * 8 + v) * 64 + n] = fmaxf(accs[nt][v] + bias[nt], 0.0f);
      }
    } else {                                  // tail tile (at most one)
#pragma unroll
      for (int nt = 0; nt < 4; ++nt) {
        const int n = nt * 16 + mrow;
#pragma unroll
        for (int v = 0; v < 8; ++v) {
          const long m = base + half * 8 + v;
          if (m < N) h[m * 64 + n] = fmaxf(accs[nt][v] + bias[nt], 0.0f);
        }
      }
    }
  }
}

// ---------------------------------------------------------------------------
// Kernel 3: score = concat(h[src], h[dst]) @ W_pred^T + b   (N_CLASSES = 2)
// Tiny-N GEMM: VALU fma per edge; h is L2-resident (25.6 MB << 192 MB L2).
// ---------------------------------------------------------------------------
__global__ void __launch_bounds__(256)
k_pred(const float* __restrict__ h,
       const long long* __restrict__ ei,
       const float* __restrict__ Wp,           // [2][128]
       const float* __restrict__ bp,           // [2]
       float* __restrict__ out,                // [E][2]
       int E) {
  int e = blockIdx.x * blockDim.x + threadIdx.x;
  if (e >= E) return;
  const float4* hs = (const float4*)(h + (long)(int)ei[e] * 64);
  const float4* hd = (const float4*)(h + (long)(int)ei[(long)E + e] * 64);
  float s0 = bp[0], s1 = bp[1];
#pragma unroll
  for (int q = 0; q < 16; ++q) {              // 4 floats per iter, K = 64
    const float4 a = hs[q];
    const float4 b = hd[q];
    const int k = q * 4;
#pragma unroll
    for (int j = 0; j < 4; ++j) {
      const float aj = (&a.x)[j], bj = (&b.x)[j];
      s0 = fmaf(aj, Wp[k + j],       fmaf(bj, Wp[64 + k + j],  s0));
      s1 = fmaf(aj, Wp[128 + k + j], fmaf(bj, Wp[192 + k + j], s1));
    }
  }
  out[(long)e * 2]     = s0;
  out[(long)e * 2 + 1] = s1;
}

// ---------------------------------------------------------------------------
extern "C" void kernel_launch(void* const* d_in, const int* in_sizes, int n_in,
                              void* d_out, int out_size, void* d_ws,
                              size_t ws_size, hipStream_t stream) {
  const float*     x  = (const float*)d_in[0];
  const long long* ei = (const long long*)d_in[1];
  const float*     ea = (const float*)d_in[2];
  const float*     Wm = (const float*)d_in[3];
  const float*     bm = (const float*)d_in[4];
  const float*     Wa = (const float*)d_in[5];
  const float*     ba = (const float*)d_in[6];
  const float*     Wp = (const float*)d_in[7];
  const float*     bp = (const float*)d_in[8];

  const int N = in_sizes[0] / 64;   // nodes
  const int E = in_sizes[1] / 2;    // edges (edge_index is [2][E])

  // Workspace: sums[N*64] | cnt[N] | h[N*64]  (~51.6 MB)
  float* sums = (float*)d_ws;
  float* cnt  = sums + (size_t)N * 64;
  float* h    = cnt + N;
  float* out  = (float*)d_out;

  {  // zero sums + cnt (contiguous)
    const long n  = (long)N * 65;
    const long n4 = n >> 2;
    const int  tail = (int)(n & 3);
    k_zero<<<(int)((n4 + 255) / 256), 256, 0, stream>>>(
        (float4*)sums, n4, sums + n4 * 4, tail);
  }
  {  // message GEMM + scatter-add (WMMA bf16, grid-stride tiles)
    const int tiles  = (E + 15) / 16;
    int blocks = (tiles + 7) / 8;
    if (blocks > 1024) blocks = 1024;         // ~10 tiles/wave, staging amortized
    k_msg_scatter<<<blocks, 256, 0, stream>>>(x, ei, ea, Wm, bm, sums, cnt, E);
  }
  {  // node update GEMM + relu (WMMA bf16, grid-stride tiles)
    const int tiles  = (N + 15) / 16;
    int blocks = (tiles + 7) / 8;
    if (blocks > 512) blocks = 512;
    k_apply<<<blocks, 256, 0, stream>>>(x, sums, cnt, Wa, ba, h, N);
  }
  {  // edge predictor
    k_pred<<<(E + 255) / 256, 256, 0, stream>>>(h, ei, Wp, bp, out, E);
  }
}
